// PoseNet_38895223832813
// MI455X (gfx1250) — matrix operand order repaired and checked by reading, compiled, not run on hardware
//
#include <hip/hip_runtime.h>
#include <hip/hip_bf16.h>

// ---------------------------------------------------------------------------
// Types
// ---------------------------------------------------------------------------
typedef __attribute__((ext_vector_type(16))) __bf16 v16bf;
typedef __attribute__((ext_vector_type(8)))  float  v8f;
typedef __attribute__((ext_vector_type(4)))  unsigned int u32x4;
typedef __attribute__((ext_vector_type(4)))  float  f32x4;

union AFrag { u32x4 u[2]; v16bf v; };

#define GN_EPS 1e-5f

// ---------------------------------------------------------------------------
// LDS layout (dynamic shared memory), bytes. All offsets 16B aligned.
// ---------------------------------------------------------------------------
constexpr int OFF_AC  = 0;                         // bf16 16x1544 (cloud) / 16x520 (text)
constexpr int OFF_H32 = OFF_AC  + 16 * 1544 * 2;   // f32  16x516 / 16x260
constexpr int OFF_ABF = OFF_H32 + 16 * 516  * 4;   // bf16 16x520 / 16x264
constexpr int OFF_H2  = OFF_ABF + 16 * 520  * 2;   // bf16 16x136
constexpr int OFF_C0  = OFF_H2  + 16 * 136  * 2;   // f32  16x32
constexpr int OFF_T0  = OFF_C0  + 16 * 32   * 4;   // f32  16x32
constexpr int OFF_SUB = OFF_T0  + 16 * 32   * 4;   // f32  6x16x16
constexpr int OFF_MSK = OFF_SUB + 6 * 256   * 4;   // f32  16
constexpr int SMEM_BYTES = OFF_MSK + 64;           // 113728

// ---------------------------------------------------------------------------
// Device helpers
// ---------------------------------------------------------------------------
__device__ __forceinline__ float gelu_f(float x) {
  return 0.5f * x * (1.0f + erff(x * 0.70710678118654752f));
}

// NT adjacent 16x16 output tiles of D = A(16xK) * W(KxN), K = 32*kc_count.
// A fragment is loaded ONCE per K-step and reused for all NT column tiles.
// A in LDS bf16 row-major (stride astride elems), W in global bf16 row-major.
template <int NT>
__device__ __forceinline__ void wmma_tiles(const __bf16* __restrict__ A_lds, int astride,
                                           const __bf16* __restrict__ W, int N,
                                           int kc_count, int nt0, int lane, v8f* accs) {
#pragma unroll
  for (int j = 0; j < NT; ++j)
    accs[j] = v8f{0.f, 0.f, 0.f, 0.f, 0.f, 0.f, 0.f, 0.f};
  const int row = lane & 15;
  const int kb  = (lane >> 4) << 3;      // 0 or 8 (K sub-block per half-wave)
  const __bf16* ap = A_lds + row * astride + kb;
  const __bf16* wp = W + (size_t)kb * N + nt0 * 16 + row;
  for (int kc = 0; kc < kc_count; ++kc) {
    // A frag: elems 0..7 = K kb..kb+7, elems 8..15 = K kb+16..kb+23 (ISA 7.12.2)
    AFrag a;
    a.u[0] = *(const u32x4*)(ap);
    a.u[1] = *(const u32x4*)(ap + 16);
    __builtin_prefetch(wp + (size_t)32 * N, 0, 1);   // global_prefetch_b8
#pragma unroll
    for (int j = 0; j < NT; ++j) {
      // B frag: lane&15 = N, elems 0..7 = K kb..kb+7, elems 8..15 = K kb+16..kb+23
      AFrag b;
#pragma unroll
      for (int s = 0; s < 8; ++s) {
        b.v[s]     = wp[(size_t)s * N + j * 16];
        b.v[s + 8] = wp[(size_t)(16 + s) * N + j * 16];
      }
      accs[j] = __builtin_amdgcn_wmma_f32_16x16x32_bf16(false, a.v, false, b.v,
                                                        (short)0, accs[j], false, false);
    }
    ap += 32;
    wp += (size_t)32 * N;
  }
}

// D-frag layout: VGPR i -> M = i + 8*(lane>=16), N = lane&15 (ISA 7.12.2)
__device__ __forceinline__ void store_f32(v8f acc, float* out, int ostride,
                                          const float* bias, int nt, int lane) {
  const int col = nt * 16 + (lane & 15);
  const int rb  = (lane >> 4) * 8;
  const float bb = bias[col];
#pragma unroll
  for (int i = 0; i < 8; ++i) out[(rb + i) * ostride + col] = acc[i] + bb;
}

__device__ __forceinline__ void store_gelu_bf16(v8f acc, __bf16* out, int ostride,
                                                const float* bias, int nt, int lane) {
  const int col = nt * 16 + (lane & 15);
  const int rb  = (lane >> 4) * 8;
  const float bb = bias[col];
#pragma unroll
  for (int i = 0; i < 8; ++i)
    out[(rb + i) * ostride + col] = (__bf16)gelu_f(acc[i] + bb);
}

// ------------------------- scalar-tail helpers -----------------------------
__device__ __forceinline__ void lin_s(const float* W, const float* b, const float* x,
                                      int fi, int fo, float* y) {
  for (int o = 0; o < fo; ++o) {
    float s = b[o];
    for (int i = 0; i < fi; ++i) s += x[i] * W[i * fo + o];
    y[o] = s;
  }
}

__device__ __forceinline__ void seq2_s(const float* W1, const float* b1, int fi, int h,
                                       const float* W2, const float* b2, int fo,
                                       const float* x, float* y) {
  float t[16];
  lin_s(W1, b1, x, fi, h, t);
  for (int i = 0; i < h; ++i) t[i] = gelu_f(t[i]);
  lin_s(W2, b2, t, h, fo, y);
}

// groupnorm(4 groups of 4 over 16) + exact gelu
__device__ __forceinline__ void gn4_gelu(const float* h, const float* gamma,
                                         const float* beta, float* q) {
  for (int gp = 0; gp < 4; ++gp) {
    float m = 0.f, v = 0.f;
    for (int k = 0; k < 4; ++k) { float x = h[gp * 4 + k]; m += x; v += x * x; }
    m *= 0.25f;
    v = v * 0.25f - m * m;
    const float inv = rsqrtf(v + GN_EPS);
    for (int k = 0; k < 4; ++k) {
      const int j = gp * 4 + k;
      q[j] = gelu_f((h[j] - m) * inv * gamma[j] + beta[j]);
    }
  }
}

// ---------------------------------------------------------------------------
// Parameter pointer table: d_in[] in setup_inputs() order, params flattened in
// jax pytree order (sorted dict keys). 5 inputs + 114 param leaves = 119.
// ---------------------------------------------------------------------------
struct PtrTab { const void* p[119]; };
#define PF(i) ((const float*)pt.p[(i)])

// ---------------------------------------------------------------------------
// fp32 -> bf16 weight conversion (prologue)
// ---------------------------------------------------------------------------
__global__ void cvt_bf16(const float* __restrict__ s, __bf16* __restrict__ d, int n) {
  int i = blockIdx.x * 256 + threadIdx.x;
  if (i < n) d[i] = (__bf16)s[i];
}

// ---------------------------------------------------------------------------
// Fused forward: one workgroup (8 wave32) per 16 rows.
// ---------------------------------------------------------------------------
__global__ __launch_bounds__(256) void posenet_fused(
    PtrTab pt,
    const __bf16* __restrict__ Wc1, const __bf16* __restrict__ Wc2,
    const __bf16* __restrict__ Wc3, const __bf16* __restrict__ Wt1,
    const __bf16* __restrict__ Wt2, const __bf16* __restrict__ Wt3,
    const __bf16* __restrict__ Wsub, float* __restrict__ Out) {
  extern __shared__ char smem[];
  __bf16* Ac    = (__bf16*)(smem + OFF_AC);
  float*  H32   = (float*)(smem + OFF_H32);
  __bf16* Abf   = (__bf16*)(smem + OFF_ABF);
  __bf16* H2bf  = (__bf16*)(smem + OFF_H2);
  float*  C0    = (float*)(smem + OFF_C0);
  float*  T0    = (float*)(smem + OFF_T0);
  float*  SUBO  = (float*)(smem + OFF_SUB);
  float*  RMASK = (float*)(smem + OFF_MSK);

  const int t    = threadIdx.x;
  const int lane = t & 31;
  const int wave = t >> 5;
  const int row0 = blockIdx.x * 16;

  const float* Gin = PF(0);
  const float* Cin = PF(1);
  const float* Tin = PF(2);
  const float* Msk = PF(3);
  const float* Tt  = PF(4);

  if (t < 16) RMASK[t] = Msk[row0 + t];
  __syncthreads();

  // ============================ CLOUD branch ===============================
  // stage masked c (16 x 1536) as bf16, LDS stride 1544
  {
    const f32x4* cv = (const f32x4*)(Cin + (size_t)row0 * 1536);
    for (int idx = t; idx < 16 * 384; idx += 256) {
      const int r = idx / 384, c4 = idx % 384;
      f32x4 v = cv[(size_t)r * 384 + c4];
      const float m = RMASK[r];
      __bf16* d = Ac + r * 1544 + c4 * 4;
      d[0] = (__bf16)(v.x * m); d[1] = (__bf16)(v.y * m);
      d[2] = (__bf16)(v.z * m); d[3] = (__bf16)(v.w * m);
    }
  }
  __syncthreads();

  // GEMM1: 1536 -> 512 (each wave: 4 column tiles in one K-sweep, 48 K-steps)
  {
    v8f accs[4];
    wmma_tiles<4>(Ac, 1544, Wc1, 512, 48, wave * 4, lane, accs);
#pragma unroll
    for (int j = 0; j < 4; ++j)
      store_f32(accs[j], H32, 516, PF(20), wave * 4 + j, lane);
  }
  __syncthreads();

  // GroupNorm(8 groups of 64) + gelu -> bf16
  if (t < 128) {
    const int r = t >> 3, gp = t & 7;
    const float* h = H32 + r * 516 + gp * 64;
    float m = 0.f, v = 0.f;
    for (int k = 0; k < 64; ++k) { float x = h[k]; m += x; v += x * x; }
    m *= (1.f / 64.f);
    v = v * (1.f / 64.f) - m * m;
    const float inv = rsqrtf(v + GN_EPS);
    const float* gmm = PF(18) + gp * 64;
    const float* bet = PF(17) + gp * 64;
    __bf16* d = Abf + r * 520 + gp * 64;
    for (int k = 0; k < 64; ++k)
      d[k] = (__bf16)gelu_f((h[k] - m) * inv * gmm[k] + bet[k]);
  }
  __syncthreads();

  // GEMM2: 512 -> 128 (+gelu)
  {
    v8f accs[1];
    wmma_tiles<1>(Abf, 520, Wc2, 128, 16, wave, lane, accs);
    store_gelu_bf16(accs[0], H2bf, 136, PF(22), wave, lane);
  }
  __syncthreads();

  // GEMM3: 128 -> 32
  if (wave < 2) {
    v8f accs[1];
    wmma_tiles<1>(H2bf, 136, Wc3, 32, 4, wave, lane, accs);
    store_f32(accs[0], C0, 32, PF(24), wave, lane);
  }
  __syncthreads();

  // ============================ TEXT branch ================================
  {
    const f32x4* tv = (const f32x4*)(Tin + (size_t)row0 * 512);
    for (int idx = t; idx < 16 * 128; idx += 256) {
      const int r = idx / 128, c4 = idx % 128;
      f32x4 v = tv[(size_t)r * 128 + c4];
      const float m = RMASK[r];
      __bf16* d = Ac + r * 520 + c4 * 4;
      d[0] = (__bf16)(v.x * m); d[1] = (__bf16)(v.y * m);
      d[2] = (__bf16)(v.z * m); d[3] = (__bf16)(v.w * m);
    }
  }
  __syncthreads();

  // GEMM1: 512 -> 256 (each wave: 2 column tiles in one K-sweep)
  {
    v8f accs[2];
    wmma_tiles<2>(Ac, 520, Wt1, 256, 16, wave * 2, lane, accs);
#pragma unroll
    for (int j = 0; j < 2; ++j)
      store_f32(accs[j], H32, 260, PF(72), wave * 2 + j, lane);
  }
  __syncthreads();

  // GroupNorm(8 groups of 32) + gelu
  if (t < 128) {
    const int r = t >> 3, gp = t & 7;
    const float* h = H32 + r * 260 + gp * 32;
    float m = 0.f, v = 0.f;
    for (int k = 0; k < 32; ++k) { float x = h[k]; m += x; v += x * x; }
    m *= (1.f / 32.f);
    v = v * (1.f / 32.f) - m * m;
    const float inv = rsqrtf(v + GN_EPS);
    const float* gmm = PF(70) + gp * 32;
    const float* bet = PF(69) + gp * 32;
    __bf16* d = Abf + r * 264 + gp * 32;
    for (int k = 0; k < 32; ++k)
      d[k] = (__bf16)gelu_f((h[k] - m) * inv * gmm[k] + bet[k]);
  }
  __syncthreads();

  // GEMM2: 256 -> 128 (+gelu)
  {
    v8f accs[1];
    wmma_tiles<1>(Abf, 264, Wt2, 128, 8, wave, lane, accs);
    store_gelu_bf16(accs[0], H2bf, 136, PF(74), wave, lane);
  }
  __syncthreads();

  // GEMM3: 128 -> 32
  if (wave < 2) {
    v8f accs[1];
    wmma_tiles<1>(H2bf, 136, Wt3, 32, 4, wave, lane, accs);
    store_f32(accs[0], T0, 32, PF(76), wave, lane);
  }
  __syncthreads();

  // ====================== subnet l1 GEMMs (6x 16x16x32) ====================
  if (wave < 6) {
    const float* Ain = (wave < 3) ? C0 : T0;
    const __bf16* Ws = Wsub + wave * 512;
    AFrag a, b;
    const int row = lane & 15, kb = (lane >> 4) << 3;
#pragma unroll
    for (int j = 0; j < 8; ++j) {
      a.v[j]     = (__bf16)Ain[row * 32 + kb + j];
      a.v[j + 8] = (__bf16)Ain[row * 32 + kb + 16 + j];
    }
#pragma unroll
    for (int j = 0; j < 8; ++j) {
      b.v[j]     = Ws[(kb + j) * 16 + row];
      b.v[j + 8] = Ws[(kb + 16 + j) * 16 + row];
    }
    v8f acc = {0.f, 0.f, 0.f, 0.f, 0.f, 0.f, 0.f, 0.f};
    acc = __builtin_amdgcn_wmma_f32_16x16x32_bf16(false, a.v, false, b.v,
                                                  (short)0, acc, false, false);
    const int subBias[6] = {28, 34, 40, 80, 86, 92};   // l1.b of {c1,c2,c3,t1,t2,t3}
    store_f32(acc, SUBO + wave * 256, 16, PF(subBias[wave]), 0, lane);
  }
  __syncthreads();

  // =========================== scalar tail =================================
  if (t < 16) {
    const int r = t;
    const size_t R = (size_t)row0 + r;
    float g7[7];
    for (int j = 0; j < 7; ++j) g7[j] = Gin[R * 7 + j];
    const float tt = Tt[R];

    float tm1[2], tm2[4], tm3[6];
    seq2_s(PF(95),  PF(96),  1, 2, PF(97),  PF(98),  2, &tt, tm1);
    seq2_s(PF(99),  PF(100), 1, 4, PF(101), PF(102), 4, &tt, tm2);
    seq2_s(PF(103), PF(104), 1, 6, PF(105), PF(106), 6, &tt, tm3);

    float gd1[6], gd2[4], gd3[2];
    seq2_s(PF(43), PF(44), 7, 6, PF(45), PF(46), 6, g7,  gd1);
    seq2_s(PF(47), PF(48), 6, 4, PF(49), PF(50), 4, gd1, gd2);
    seq2_s(PF(51), PF(52), 4, 2, PF(53), PF(54), 2, gd2, gd3);

    // subnet tails: gn(4) + gelu + l2
    float c1[2], c2[4], c3[6], t1[2], t2[4], t3[6];
    {
      const int gI[6]  = {26, 32, 38, 78, 84, 90};
      const int bI[6]  = {25, 31, 37, 77, 83, 89};
      const int wI[6]  = {29, 35, 41, 81, 87, 93};
      const int b2I[6] = {30, 36, 42, 82, 88, 94};
      const int foI[6] = {2, 4, 6, 2, 4, 6};
      float* outs[6] = {c1, c2, c3, t1, t2, t3};
      for (int s = 0; s < 6; ++s) {
        float h[16], q[16];
        for (int k = 0; k < 16; ++k) h[k] = SUBO[s * 256 + r * 16 + k];
        gn4_gelu(h, PF(gI[s]), PF(bI[s]), q);
        lin_s(PF(wI[s]), PF(b2I[s]), q, 16, foI[s], outs[s]);
      }
    }

    // ---- level 1 ----
    float x1c[12] = {c1[0], c1[1], g7[0], g7[1], g7[2], g7[3], g7[4], g7[5],
                     tm1[0], tm1[1], t1[0], t1[1]};
    float c1i[2];
    seq2_s(PF(5), PF(6), 12, 16, PF(7), PF(8), 2, x1c, c1i);
    float x1t[13] = {t1[0], t1[1], g7[0], g7[1], g7[2], g7[3], g7[4], g7[5],
                     g7[6], tm1[0], tm1[1], c1[0], c1[1]};
    float h16[16], q16[16], t1i[2];
    lin_s(PF(57), PF(58), x1t, 13, 16, h16);
    gn4_gelu(h16, PF(56), PF(55), q16);
    lin_s(PF(59), PF(60), q16, 16, 2, t1i);
    float ct1[2];
    for (int j = 0; j < 2; ++j) {
      const float m = fmaxf(c1i[j], t1i[j]);
      const float e0 = expf(c1i[j] - m), e1 = expf(t1i[j] - m);
      const float w0 = e0 / (e0 + e1);
      ct1[j] = c1[j] * w0 + t1[j] * (1.f - w0);
    }
    float u1in[6] = {gd3[0] * ct1[0] + tm1[0], gd3[1] * ct1[1] + tm1[1],
                     gd2[0], gd2[1], gd2[2], gd2[3]};
    float u1[4];
    seq2_s(PF(107), PF(108), 6, 4, PF(109), PF(110), 4, u1in, u1);

    // ---- level 2 ----
    float x2c[19], x2t[19];
    {
      int k = 0;
      for (int j = 0; j < 4; ++j) x2c[k++] = c2[j];
      for (int j = 0; j < 7; ++j) x2c[k++] = g7[j];
      for (int j = 0; j < 4; ++j) x2c[k++] = tm2[j];
      for (int j = 0; j < 4; ++j) x2c[k++] = t2[j];
      k = 0;
      for (int j = 0; j < 4; ++j) x2t[k++] = t2[j];
      for (int j = 0; j < 7; ++j) x2t[k++] = g7[j];
      for (int j = 0; j < 4; ++j) x2t[k++] = tm2[j];
      for (int j = 0; j < 4; ++j) x2t[k++] = c2[j];
    }
    float c2i[4], t2i[4];
    seq2_s(PF(9),  PF(10), 19, 4, PF(11), PF(12), 4, x2c, c2i);
    seq2_s(PF(61), PF(62), 19, 4, PF(63), PF(64), 4, x2t, t2i);
    float ct2[4];
    for (int j = 0; j < 4; ++j) {
      const float m = fmaxf(c2i[j], t2i[j]);
      const float e0 = expf(c2i[j] - m), e1 = expf(t2i[j] - m);
      const float w0 = e0 / (e0 + e1);
      ct2[j] = c2[j] * w0 + t2[j] * (1.f - w0);
    }
    float u2in[10] = {u1[0] * ct2[0] + tm2[0], u1[1] * ct2[1] + tm2[1],
                      u1[2] * ct2[2] + tm2[2], u1[3] * ct2[3] + tm2[3],
                      gd1[0], gd1[1], gd1[2], gd1[3], gd1[4], gd1[5]};
    float u2[6];
    seq2_s(PF(111), PF(112), 10, 6, PF(113), PF(114), 6, u2in, u2);

    // ---- level 3 ----
    float x3c[25], x3t[25];
    {
      int k = 0;
      for (int j = 0; j < 6; ++j) x3c[k++] = c3[j];
      for (int j = 0; j < 7; ++j) x3c[k++] = g7[j];
      for (int j = 0; j < 6; ++j) x3c[k++] = tm3[j];
      for (int j = 0; j < 6; ++j) x3c[k++] = t3[j];
      k = 0;
      for (int j = 0; j < 6; ++j) x3t[k++] = t3[j];
      for (int j = 0; j < 7; ++j) x3t[k++] = g7[j];
      for (int j = 0; j < 6; ++j) x3t[k++] = tm3[j];
      for (int j = 0; j < 6; ++j) x3t[k++] = c3[j];
    }
    float c3i[6], t3i[6];
    seq2_s(PF(13), PF(14), 25, 6, PF(15), PF(16), 6, x3c, c3i);
    seq2_s(PF(65), PF(66), 25, 6, PF(67), PF(68), 6, x3t, t3i);
    float ct3[6];
    for (int j = 0; j < 6; ++j) {
      const float m = fmaxf(c3i[j], t3i[j]);
      const float e0 = expf(c3i[j] - m), e1 = expf(t3i[j] - m);
      const float w0 = e0 / (e0 + e1);
      ct3[j] = c3[j] * w0 + t3[j] * (1.f - w0);
    }
    float u3in[13] = {u2[0] * ct3[0] + tm3[0], u2[1] * ct3[1] + tm3[1],
                      u2[2] * ct3[2] + tm3[2], u2[3] * ct3[3] + tm3[3],
                      u2[4] * ct3[4] + tm3[4], u2[5] * ct3[5] + tm3[5],
                      g7[0], g7[1], g7[2], g7[3], g7[4], g7[5], g7[6]};
    float out7[7];
    seq2_s(PF(115), PF(116), 13, 7, PF(117), PF(118), 7, u3in, out7);
    for (int j = 0; j < 7; ++j) Out[R * 7 + j] = out7[j];
  }
}

// ---------------------------------------------------------------------------
// Host launch
// ---------------------------------------------------------------------------
extern "C" void kernel_launch(void* const* d_in, const int* in_sizes, int n_in,
                              void* d_out, int out_size, void* d_ws, size_t ws_size,
                              hipStream_t stream) {
  (void)in_sizes; (void)out_size; (void)ws_size;
  const int B = 65536;

  // bf16 weight cache in workspace
  char* ws = (char*)d_ws;
  __bf16* Wc1  = (__bf16*)(ws + 0);        // 1536x512
  __bf16* Wc2  = (__bf16*)(ws + 1572864);  // 512x128
  __bf16* Wc3  = (__bf16*)(ws + 1703936);  // 128x32
  __bf16* Wt1  = (__bf16*)(ws + 1712128);  // 512x256
  __bf16* Wt2  = (__bf16*)(ws + 1974272);  // 256x128
  __bf16* Wt3  = (__bf16*)(ws + 2039808);  // 128x32
  __bf16* Wsub = (__bf16*)(ws + 2048000);  // 6 x 32x16

  struct Cvt { int idx; __bf16* dst; int n; };
  const Cvt cv[12] = {
      {19, Wc1, 786432}, {21, Wc2, 65536}, {23, Wc3, 4096},
      {71, Wt1, 131072}, {73, Wt2, 32768}, {75, Wt3, 4096},
      {27, Wsub + 0,    512}, {33, Wsub + 512,  512}, {39, Wsub + 1024, 512},
      {79, Wsub + 1536, 512}, {85, Wsub + 2048, 512}, {91, Wsub + 2560, 512}};
  for (int i = 0; i < 12; ++i)
    cvt_bf16<<<(cv[i].n + 255) / 256, 256, 0, stream>>>(
        (const float*)d_in[cv[i].idx], cv[i].dst, cv[i].n);

  PtrTab pt;
  for (int i = 0; i < 119; ++i) pt.p[i] = (i < n_in) ? d_in[i] : nullptr;

  hipFuncSetAttribute(reinterpret_cast<const void*>(posenet_fused),
                      hipFuncAttributeMaxDynamicSharedMemorySize, SMEM_BYTES);
  posenet_fused<<<B / 16, 256, SMEM_BYTES, stream>>>(
      pt, Wc1, Wc2, Wc3, Wt1, Wt2, Wt3, Wsub, (float*)d_out);
}
